// MultiHopMGAT_40785009443441
// MI455X (gfx1250) — matrix operand order repaired and checked by reading, not compiled, not run on hardware
//
#include <hip/hip_runtime.h>
#include <hip/hip_bf16.h>
#include <stdint.h>

#define NN      2048
#define INCH    512
#define HIDC    64
#define NHEADS  4
#define OUTC    64
#define EEDGES  32768
#define BETA_   0.5f
#define NSLOPE  0.2f

typedef _Float16 half_t;
typedef __attribute__((ext_vector_type(16))) _Float16 v16h;
typedef __attribute__((ext_vector_type(8)))  _Float16 v8h;
typedef __attribute__((ext_vector_type(8)))  float    v8f;
typedef __attribute__((ext_vector_type(4)))  int      v4i;

#if defined(__AMDGCN__) && __has_builtin(__builtin_amdgcn_global_load_async_to_lds_b128) && \
    __has_builtin(__builtin_amdgcn_s_wait_asynccnt)
#define USE_ASYNC_LDS 1
#define AS1 __attribute__((address_space(1)))
#define AS3 __attribute__((address_space(3)))
#endif

__device__ __forceinline__ float leaky_(float v) { return v >= 0.0f ? v : NSLOPE * v; }

// ---------------------------------------------------------------- utilities
__global__ void k_fill_h(half_t* __restrict__ p, int nElt) {
  int i = blockIdx.x * blockDim.x + threadIdx.x;
  if (i < nElt) p[i] = (half_t)0.0f;
}

__global__ void k_scatter_edges(const int* __restrict__ ei, half_t* __restrict__ A,
                                int n, int e) {
  int i = blockIdx.x * blockDim.x + threadIdx.x;
  if (i < e) {
    int s = ei[i];
    int d = ei[e + i];
    A[(size_t)s * n + d] = (half_t)1.0f;   // duplicates write same value: benign
  }
}

__global__ void k_f32_to_f16(const float* __restrict__ s, half_t* __restrict__ d, int nElt) {
  int i = blockIdx.x * blockDim.x + threadIdx.x;
  if (i < nElt) d[i] = (half_t)s[i];
}

// ------------------------------------------------- WMMA f16 GEMM: C = A @ B
// A: MxK f16 row-major, B: KxN f16 row-major, C: MxN f32.
// Block = 256 threads (8 waves). Block tile 128x64, BK = 32.
// Wave w owns rows [w*16, w*16+16) x all 64 cols -> 4 16x16 WMMA accumulators.
// A tile staged via gfx1250 async global->LDS b128 (ASYNCcnt) when available;
// B tile staged TRANSPOSED (Bt[n][k], padded stride) so every lane's 16
// B-halves are two contiguous ds_load_b128 -- no strided 16-bit gathers.
#define BTS 40   // padded Bt row stride in halves (80B, 16B-aligned)
__global__ __launch_bounds__(256)
void k_gemm_f16_wmma(const half_t* __restrict__ A, const half_t* __restrict__ B,
                     float* __restrict__ C, int M, int Nn, int K) {
  __shared__ __align__(16) half_t As[128 * 32];   // [row][k]
  __shared__ __align__(16) half_t Bt[64 * BTS];   // [col][k] (transposed)

  const int tid   = threadIdx.x;
  const int wave  = tid >> 5;         // 0..7 -> row slab
  const int lane  = tid & 31;
  const int bm    = blockIdx.y * 128;
  const int bn    = blockIdx.x * 64;
  const int row16 = lane & 15;
  const int hi    = lane >> 4;        // half-wave select
  const int kb    = hi * 8;           // A K-base within 16
  const int kbB   = hi * 16;          // unused; B reads use kb, same k order as A

  // cooperative tile-load coordinates (16B per thread per load)
  const int a_r = tid >> 2;           // 0..63 (plus +64 for second half)
  const int a_c = (tid & 3) << 3;     // 0,8,16,24
  const int b_r = tid >> 3;           // k: 0..31
  const int b_c = (tid & 7) << 3;     // n: 0..56

  v8f acc0 = {}, acc1 = {}, acc2 = {}, acc3 = {};

  for (int k0 = 0; k0 < K; k0 += 32) {
    if (k0 + 32 < K) {  // gfx1250 global_prefetch_b8 of next K tiles
      __builtin_prefetch(&A[(size_t)(bm + a_r) * K + (k0 + 32) + a_c], 0, 1);
      __builtin_prefetch(&B[(size_t)(k0 + 32 + b_r) * Nn + bn + b_c], 0, 1);
    }

    // ---- stage A tile (async global->LDS, tracked by ASYNCcnt) ----
#ifdef USE_ASYNC_LDS
    __builtin_amdgcn_global_load_async_to_lds_b128(
        (AS1 v4i*)&A[(size_t)(bm + a_r) * K + k0 + a_c],
        (AS3 v4i*)&As[a_r * 32 + a_c], 0, 0);
    __builtin_amdgcn_global_load_async_to_lds_b128(
        (AS1 v4i*)&A[(size_t)(bm + a_r + 64) * K + k0 + a_c],
        (AS3 v4i*)&As[(a_r + 64) * 32 + a_c], 0, 0);
#else
    *(uint4*)&As[a_r * 32 + a_c] =
        *(const uint4*)&A[(size_t)(bm + a_r) * K + k0 + a_c];
    *(uint4*)&As[(a_r + 64) * 32 + a_c] =
        *(const uint4*)&A[(size_t)(bm + a_r + 64) * K + k0 + a_c];
#endif

    // ---- stage B tile transposed: Bt[n][k] ----
    {
      v8h bv = *(const v8h*)&B[(size_t)(k0 + b_r) * Nn + bn + b_c];
      #pragma unroll
      for (int j = 0; j < 8; ++j) Bt[(b_c + j) * BTS + b_r] = bv[j];
    }

#ifdef USE_ASYNC_LDS
    __builtin_amdgcn_s_wait_asynccnt(0);
#endif
    __syncthreads();

    // ---- A fragment: two contiguous b128 reads + shuffle ----
    const half_t* ap = &As[(wave * 16 + row16) * 32];
    v8h a_lo = *(const v8h*)&ap[kb];
    v8h a_hi = *(const v8h*)&ap[16 + kb];
    v16h af = __builtin_shufflevector(a_lo, a_hi, 0, 1, 2, 3, 4, 5, 6, 7,
                                      8, 9, 10, 11, 12, 13, 14, 15);

    // ---- 4 B fragments (one per 16-col tile), contiguous reads ----
    #pragma unroll
    for (int t = 0; t < 4; ++t) {
      const half_t* bp = &Bt[(t * 16 + row16) * BTS];
      v8h b_lo = *(const v8h*)&bp[kb];
      v8h b_hi = *(const v8h*)&bp[16 + kb];
      v16h bf = __builtin_shufflevector(b_lo, b_hi, 0, 1, 2, 3, 4, 5, 6, 7,
                                        8, 9, 10, 11, 12, 13, 14, 15);
      if (t == 0) acc0 = __builtin_amdgcn_wmma_f32_16x16x32_f16(false, af, false, bf, (short)0, acc0, false, false);
      if (t == 1) acc1 = __builtin_amdgcn_wmma_f32_16x16x32_f16(false, af, false, bf, (short)0, acc1, false, false);
      if (t == 2) acc2 = __builtin_amdgcn_wmma_f32_16x16x32_f16(false, af, false, bf, (short)0, acc2, false, false);
      if (t == 3) acc3 = __builtin_amdgcn_wmma_f32_16x16x32_f16(false, af, false, bf, (short)0, acc3, false, false);
    }
    __syncthreads();
  }

  // C layout: VGPR r -> M = r (+8 for upper half-wave), N = lane&15
  const int crow = bm + wave * 16 + hi * 8;
  #pragma unroll
  for (int r = 0; r < 8; ++r) {
    C[(size_t)(crow + r) * Nn + bn + row16]      = acc0[r];
    C[(size_t)(crow + r) * Nn + bn + 16 + row16] = acc1[r];
    C[(size_t)(crow + r) * Nn + bn + 32 + row16] = acc2[r];
    C[(size_t)(crow + r) * Nn + bn + 48 + row16] = acc3[r];
  }
}

// ------------------------------------------------- motif row-normalization
__global__ __launch_bounds__(256)
void k_motif_norm(float* __restrict__ Mf, int n) {
  __shared__ float red[256];
  const int row = blockIdx.x;
  const int tid = threadIdx.x;
  float s = 0.0f;
  for (int j = tid; j < n; j += 256) s += Mf[(size_t)row * n + j];
  red[tid] = s;
  __syncthreads();
  for (int off = 128; off > 0; off >>= 1) {
    if (tid < off) red[tid] += red[tid + off];
    __syncthreads();
  }
  const float inv = 1.0f / fmaxf(red[0], 1.0f);
  for (int j = tid; j < n; j += 256) Mf[(size_t)row * n + j] *= inv;
}

// ------------------------------------------------- attention score vectors
__global__ void k_scores(const float* __restrict__ hopx, const float* __restrict__ att_src,
                         const float* __restrict__ att_dst, float* __restrict__ ssrc,
                         float* __restrict__ sdst, int n, int heads, int ch) {
  const int i = blockIdx.x * blockDim.x + threadIdx.x;
  const int tot = 2 * n * heads;
  if (i >= tot) return;
  const int hop  = i / (n * heads);
  const int r    = i % (n * heads);
  const int node = r / heads;
  const int h    = r % heads;
  const float* xp = &hopx[((size_t)hop * n + node) * (heads * ch) + h * ch];
  const float* as = &att_src[(hop * heads + h) * ch];
  const float* ad = &att_dst[(hop * heads + h) * ch];
  float s1 = 0.0f, s2 = 0.0f;
  for (int c = 0; c < ch; ++c) { float v = xp[c]; s1 += v * as[c]; s2 += v * ad[c]; }
  ssrc[((size_t)hop * n + node) * heads + h] = s1;
  sdst[((size_t)hop * n + node) * heads + h] = s2;
}

// ------------------------------- fused masked dual-softmax + sparse gather
template <int HEADS_>
__global__ __launch_bounds__(256)
void k_aggregate(const half_t* __restrict__ Ah, const half_t* __restrict__ A2h,
                 const float* __restrict__ motif, const float* __restrict__ ssrc,
                 const float* __restrict__ sdst, const float* __restrict__ hopx,
                 const float* __restrict__ hop_att, float* __restrict__ out,
                 int n, int ch) {
  __shared__ unsigned short sh_idx[NN];
  __shared__ float sh_mot[NN];
  __shared__ float sh_sc[NN * HEADS_];
  __shared__ int   sh_cnt;
  __shared__ float redA[256], redB[256];
  __shared__ float gmax1[HEADS_], gmax2[HEADS_], gsum1[HEADS_], gsum2[HEADS_];

  const int d   = blockIdx.x;
  const int tid = threadIdx.x;
  constexpr int TPH = 256 / HEADS_;
  const int h  = tid / TPH;
  const int lg = tid % TPH;
  const int HC = HEADS_ * ch;

  const float a0 = hop_att[0], a1 = hop_att[1];
  const float mx = fmaxf(a0, a1);
  const float e0 = __expf(a0 - mx), e1w = __expf(a1 - mx);
  const float hw0 = e0 / (e0 + e1w), hw1 = e1w / (e0 + e1w);

  float acc = 0.0f;

  for (int hop = 0; hop < 2; ++hop) {
    if (tid == 0) sh_cnt = 0;
    __syncthreads();
    const half_t* Acol = hop ? A2h : Ah;
    for (int s = tid; s < n; s += 256) {
      const float av = (float)Acol[(size_t)s * n + d];
      if (av > 0.0f || s == d) {
        const int p = atomicAdd(&sh_cnt, 1);
        sh_idx[p] = (unsigned short)s;
        sh_mot[p] = motif[(size_t)s * n + d];
        #pragma unroll
        for (int hh = 0; hh < HEADS_; ++hh)
          sh_sc[p * HEADS_ + hh] = ssrc[((size_t)hop * n + s) * HEADS_ + hh];
      }
    }
    __syncthreads();
    const int cnt = sh_cnt;
    const float sdh = sdst[((size_t)hop * n + d) * HEADS_ + h];

    float m1 = -3.0e38f, m2 = -3.0e38f;
    for (int p = lg; p < cnt; p += TPH) {
      const float sc = sh_sc[p * HEADS_ + h] + sdh;
      m1 = fmaxf(m1, leaky_(sc));
      m2 = fmaxf(m2, leaky_(sc * sh_mot[p]));
    }
    redA[tid] = m1; redB[tid] = m2;
    __syncthreads();
    for (int off = TPH >> 1; off > 0; off >>= 1) {
      if (lg < off) {
        redA[tid] = fmaxf(redA[tid], redA[tid + off]);
        redB[tid] = fmaxf(redB[tid], redB[tid + off]);
      }
      __syncthreads();
    }
    if (lg == 0) { gmax1[h] = redA[tid]; gmax2[h] = redB[tid]; }
    __syncthreads();
    const float M1 = gmax1[h], M2 = gmax2[h];

    float s1 = 0.0f, s2 = 0.0f;
    for (int p = lg; p < cnt; p += TPH) {
      const float sc = sh_sc[p * HEADS_ + h] + sdh;
      s1 += __expf(leaky_(sc) - M1);
      s2 += __expf(leaky_(sc * sh_mot[p]) - M2);
    }
    redA[tid] = s1; redB[tid] = s2;
    __syncthreads();
    for (int off = TPH >> 1; off > 0; off >>= 1) {
      if (lg < off) { redA[tid] += redA[tid + off]; redB[tid] += redB[tid + off]; }
      __syncthreads();
    }
    if (lg == 0) { gsum1[h] = redA[tid]; gsum2[h] = redB[tid]; }
    __syncthreads();
    const float i1 = 1.0f / gsum1[h];
    const float i2 = 1.0f / gsum2[h];

    for (int p = lg; p < cnt; p += TPH) {
      const float sc = sh_sc[p * HEADS_ + h] + sdh;
      const float w = BETA_ * __expf(leaky_(sc) - M1) * i1 +
                      (1.0f - BETA_) * __expf(leaky_(sc * sh_mot[p]) - M2) * i2;
      sh_sc[p * HEADS_ + h] = w;
    }
    __syncthreads();

    if (tid < HC) {
      const int hh = tid / ch, c = tid % ch;
      float a = 0.0f;
      for (int p = 0; p < cnt; ++p) {
        const int s = sh_idx[p];
        a += sh_sc[p * HEADS_ + hh] * hopx[((size_t)hop * n + s) * HC + hh * ch + c];
      }
      acc += (hop ? hw1 : hw0) * a;
    }
    __syncthreads();
  }
  if (tid < HC) out[(size_t)d * HC + tid] = acc;
}

// ------------------------------------------------- layer-1 finalize: bias+ELU
__global__ void k_l1_final(const float* __restrict__ out1, const float* __restrict__ bias,
                           half_t* __restrict__ hh, int total, int hc) {
  const int i = blockIdx.x * blockDim.x + threadIdx.x;
  if (i < total) {
    float v = out1[i] + bias[i % hc];
    v = v > 0.0f ? v : (__expf(v) - 1.0f);
    hh[i] = (half_t)v;
  }
}

// ------------------------------------------------- layer-2 finalize: res+LN+bias
__global__ __launch_bounds__(64)
void k_l2_final(const float* __restrict__ out2, const float* __restrict__ res,
                const float* __restrict__ ln_s, const float* __restrict__ ln_b,
                const float* __restrict__ bias, float* __restrict__ y, int ch) {
  __shared__ float red[64];
  const int d = blockIdx.x, c = threadIdx.x;
  const float v = out2[(size_t)d * ch + c] + res[(size_t)d * ch + c];
  red[c] = v;
  __syncthreads();
  for (int off = 32; off > 0; off >>= 1) { if (c < off) red[c] += red[c + off]; __syncthreads(); }
  const float mu = red[0] / (float)ch;
  __syncthreads();
  const float dv = v - mu;
  red[c] = dv * dv;
  __syncthreads();
  for (int off = 32; off > 0; off >>= 1) { if (c < off) red[c] += red[c + off]; __syncthreads(); }
  const float var = red[0] / (float)ch;
  y[(size_t)d * ch + c] = dv * rsqrtf(var + 1e-5f) * ln_s[c] + ln_b[c] + bias[c];
}

// ============================================================== launcher
extern "C" void kernel_launch(void* const* d_in, const int* in_sizes, int n_in,
                              void* d_out, int out_size, void* d_ws, size_t ws_size,
                              hipStream_t stream) {
  const float* x        = (const float*)d_in[0];
  const int*   ei       = (const int*)  d_in[1];
  const float* w1       = (const float*)d_in[2];   // [2,512,256]
  const float* a1s      = (const float*)d_in[3];   // [2,4,64]
  const float* a1d      = (const float*)d_in[4];
  const float* hop1     = (const float*)d_in[5];   // [2]
  const float* b1       = (const float*)d_in[6];   // [256]
  const float* w2       = (const float*)d_in[7];   // [2,256,64]
  const float* a2s      = (const float*)d_in[8];   // [2,1,64]
  const float* a2d      = (const float*)d_in[9];
  const float* hop2     = (const float*)d_in[10];  // [2]
  const float* rw       = (const float*)d_in[11];  // [256,64]
  const float* lns      = (const float*)d_in[12];
  const float* lnb      = (const float*)d_in[13];
  const float* b2       = (const float*)d_in[14];
  float*       y        = (float*)d_out;

  char* w = (char*)d_ws;
  size_t off = 0;
  auto grab = [&](size_t bytes) { char* p = w + off; off += (bytes + 255) & ~(size_t)255; return p; };

  half_t* Ah     = (half_t*)grab((size_t)NN * NN * 2);
  half_t* A2h    = (half_t*)grab((size_t)NN * NN * 2);
  float*  motifF = (float*) grab((size_t)NN * NN * 4);
  half_t* xh     = (half_t*)grab((size_t)NN * INCH * 2);
  half_t* w1h    = (half_t*)grab((size_t)2 * INCH * 256 * 2);
  half_t* w2h    = (half_t*)grab((size_t)2 * 256 * 64 * 2);
  half_t* rwh    = (half_t*)grab((size_t)256 * 64 * 2);
  float*  hopx1  = (float*) grab((size_t)2 * NN * 256 * 4);
  float*  hopx2  = (float*) grab((size_t)2 * NN * 64 * 4);
  float*  s1s    = (float*) grab((size_t)2 * NN * NHEADS * 4);
  float*  s1d    = (float*) grab((size_t)2 * NN * NHEADS * 4);
  float*  s2s    = (float*) grab((size_t)2 * NN * 4);
  float*  s2d    = (float*) grab((size_t)2 * NN * 4);
  float*  out1   = (float*) grab((size_t)NN * 256 * 4);
  half_t* hh     = (half_t*)grab((size_t)NN * 256 * 2);
  float*  resb   = (float*) grab((size_t)NN * 64 * 4);
  float*  out2   = (float*) grab((size_t)NN * 64 * 4);

  auto cdiv = [](int a, int b) { return (a + b - 1) / b; };

  // --- build adjacency (f16: counts exact) -------------------------------
  k_fill_h<<<cdiv(NN * NN, 256), 256, 0, stream>>>(Ah, NN * NN);
  k_scatter_edges<<<cdiv(EEDGES, 256), 256, 0, stream>>>(ei, Ah, NN, EEDGES);

  // --- f32 -> f16 weight / feature conversions ---------------------------
  k_f32_to_f16<<<cdiv(NN * INCH, 256), 256, 0, stream>>>(x, xh, NN * INCH);
  k_f32_to_f16<<<cdiv(2 * INCH * 256, 256), 256, 0, stream>>>(w1, w1h, 2 * INCH * 256);
  k_f32_to_f16<<<cdiv(2 * 256 * 64, 256), 256, 0, stream>>>(w2, w2h, 2 * 256 * 64);
  k_f32_to_f16<<<cdiv(256 * 64, 256), 256, 0, stream>>>(rw, rwh, 256 * 64);

  // --- adjacency powers via WMMA -----------------------------------------
  { dim3 g(NN / 64, NN / 128);
    k_gemm_f16_wmma<<<g, 256, 0, stream>>>(Ah, Ah, motifF, NN, NN, NN); }       // A^2 (f32)
  k_f32_to_f16<<<cdiv(NN * NN, 256), 256, 0, stream>>>(motifF, A2h, NN * NN);   // exact ints
  { dim3 g(NN / 64, NN / 128);
    k_gemm_f16_wmma<<<g, 256, 0, stream>>>(A2h, Ah, motifF, NN, NN, NN); }      // A^3
  k_motif_norm<<<NN, 256, 0, stream>>>(motifF, NN);

  // --- layer 1 ------------------------------------------------------------
  for (int hop = 0; hop < 2; ++hop) {
    dim3 g(256 / 64, NN / 128);
    k_gemm_f16_wmma<<<g, 256, 0, stream>>>(xh, w1h + (size_t)hop * INCH * 256,
                                           hopx1 + (size_t)hop * NN * 256, NN, 256, INCH);
  }
  k_scores<<<cdiv(2 * NN * NHEADS, 256), 256, 0, stream>>>(hopx1, a1s, a1d, s1s, s1d,
                                                           NN, NHEADS, HIDC);
  k_aggregate<NHEADS><<<NN, 256, 0, stream>>>(Ah, A2h, motifF, s1s, s1d, hopx1, hop1,
                                              out1, NN, HIDC);
  k_l1_final<<<cdiv(NN * 256, 256), 256, 0, stream>>>(out1, b1, hh, NN * 256, 256);

  // --- layer 2 ------------------------------------------------------------
  for (int hop = 0; hop < 2; ++hop) {
    dim3 g(64 / 64, NN / 128);
    k_gemm_f16_wmma<<<g, 256, 0, stream>>>(hh, w2h + (size_t)hop * 256 * 64,
                                           hopx2 + (size_t)hop * NN * 64, NN, 64, 256);
  }
  k_scores<<<cdiv(2 * NN, 256), 256, 0, stream>>>(hopx2, a2s, a2d, s2s, s2d, NN, 1, OUTC);
  { dim3 g(64 / 64, NN / 128);
    k_gemm_f16_wmma<<<g, 256, 0, stream>>>(hh, rwh, resb, NN, 64, 256); }       // residual
  k_aggregate<1><<<NN, 256, 0, stream>>>(Ah, A2h, motifF, s2s, s2d, hopx2, hop2,
                                         out2, NN, OUTC);
  k_l2_final<<<NN, 64, 0, stream>>>(out2, resb, lns, lnb, b2, y, OUTC);
}
